// WanI2VCrossAttentionProcessor_50019189129891
// MI455X (gfx1250) — matrix operand order repaired and checked by reading, not compiled, not run on hardware
//
#include <hip/hip_runtime.h>

// ---------------------------------------------------------------------------
// MI455X (gfx1250): bf16 WMMA (v_wmma_f32_16x16x32_bf16), wave32, 8-wave WGs.
// Round 3: async global->LDS staging of K chunks (global_load_async_to_lds_b128
// + s_wait_asynccnt, double-buffered), in addition to the batched-fragment /
// software-pipelined structure from round 2.
// ---------------------------------------------------------------------------

#define USE_ASYNC_LDS 1

typedef __attribute__((ext_vector_type(16))) __bf16        v16bf;
typedef __attribute__((ext_vector_type(8)))  float         v8f;
typedef __attribute__((ext_vector_type(8)))  unsigned int  v8u;
typedef __attribute__((ext_vector_type(4)))  unsigned int  v4u;
typedef __attribute__((ext_vector_type(2)))  unsigned int  v2u;
typedef __attribute__((ext_vector_type(4)))  float         v4f;

__device__ __forceinline__ unsigned short f32_to_bf16(float f) {
    union { float f; unsigned int u; } v; v.f = f;
    unsigned int r = v.u + 0x7FFFu + ((v.u >> 16) & 1u);   // round-to-nearest-even
    return (unsigned short)(r >> 16);
}

// Build a 16-element bf16 fragment from two 16-byte chunks (LDS or global).
__device__ __forceinline__ v16bf load_frag(const unsigned short* lo,
                                           const unsigned short* hi) {
    v4u a = *(const v4u*)lo;
    v4u b = *(const v4u*)hi;
    v8u r;
    r[0] = a[0]; r[1] = a[1]; r[2] = a[2]; r[3] = a[3];
    r[4] = b[0]; r[5] = b[1]; r[6] = b[2]; r[7] = b[3];
    return __builtin_bit_cast(v16bf, r);
}

__device__ __forceinline__ v8f wmma_bf16(v16bf a, v16bf b, v8f c) {
    // (neg_a, A, neg_b, B, c_mod, C, reuse_a, reuse_b)
    return __builtin_amdgcn_wmma_f32_16x16x32_bf16(false, a, false, b,
                                                   (short)0, c, false, false);
}

// Async copy 16B global -> LDS (ASYNCcnt-tracked, no VGPR round-trip).
__device__ __forceinline__ void async_copy_b128(const void* gsrc, void* ldst) {
#if USE_ASYNC_LDS
    unsigned           dst = (unsigned)(unsigned long long)ldst;   // LDS byte offset
    unsigned long long src = (unsigned long long)gsrc;
    asm volatile("global_load_async_to_lds_b128 %0, %1, off"
                 :: "v"(dst), "v"(src) : "memory");
#else
    *(v4u*)ldst = *(const v4u*)gsrc;
#endif
}
__device__ __forceinline__ void wait_async_le4() {
#if USE_ASYNC_LDS
    asm volatile("s_wait_asynccnt 0x4" ::: "memory");
#endif
}
__device__ __forceinline__ void wait_async_0() {
#if USE_ASYNC_LDS
    asm volatile("s_wait_asynccnt 0x0" ::: "memory");
#endif
}

// ---------------------------------------------------------------------------
// Tiled GEMM:  Y[M,N](f32) = A[M,K] @ W[K,N](f32) + bias[N]
//   Tile 128x128, K-step 32, double-buffered LDS, software-pipelined.
//   8 waves: wave = (wm 0..3) x (wn 0..1); each wave: 32x64 = 2x4 accumulators.
// ---------------------------------------------------------------------------
#define TILE_M 128
#define TILE_N 128
#define TILE_K 32

template <int A_BF16>
__global__ __launch_bounds__(256)
void gemm_bias(const void* __restrict__ A, const float* __restrict__ W,
               const float* __restrict__ bias, float* __restrict__ Y,
               int M, int K, int N,
               long long aBatchStride, long long yBatchStride)
{
    __shared__ unsigned short As [2][TILE_M][TILE_K + 8];
    __shared__ unsigned short Bts[2][TILE_N][TILE_K + 8];   // transposed W tile

    const int tid  = threadIdx.x;
    const int lane = tid & 31;
    const int wave = tid >> 5;
    const int wm   = wave >> 1;          // 0..3 : 32-row band
    const int wn   = wave & 1;           // 0..1 : 64-col band
    const int lr   = lane & 15;
    const int half = lane >> 4;
    const int hk8  = half << 3;          // 0 / 8
    const int hk16 = half << 4;          // 0 / 16

    const int m0 = blockIdx.y * TILE_M;
    const int n0 = blockIdx.x * TILE_N;
    const int b  = blockIdx.z;

    const float*          Af = (const float*)A          + (size_t)b * aBatchStride;
    const unsigned short* Ah = (const unsigned short*)A + (size_t)b * aBatchStride;
    float*                Yb = Y                        + (size_t)b * yBatchStride;

    v8f acc[2][4];
    #pragma unroll
    for (int mt = 0; mt < 2; ++mt)
        #pragma unroll
        for (int nt = 0; nt < 4; ++nt)
            acc[mt][nt] = (v8f)0.0f;

    // staging registers (tile for the *next* K-step)
    v4f aF[4];
    v2u aH[4];
    v4f wR[4];

    auto gload = [&](int k0) {
        #pragma unroll
        for (int i = 0; i < 4; ++i) {
            int idx = tid + (i << 8);                 // 0..1023
            if (A_BF16) {
                int row = idx >> 3, cc = (idx & 7) << 2;
                aH[i] = (v2u)0u;
                if (m0 + row < M)
                    aH[i] = *(const v2u*)&Ah[(size_t)(m0 + row) * K + k0 + cc];
            } else {
                int row = idx >> 3, cc = (idx & 7) << 2;
                aF[i] = (v4f)0.0f;
                if (m0 + row < M)
                    aF[i] = *(const v4f*)&Af[(size_t)(m0 + row) * K + k0 + cc];
            }
            int krow = idx >> 5, cc2 = (idx & 31) << 2;
            wR[i] = *(const v4f*)&W[(size_t)(k0 + krow) * N + n0 + cc2];
        }
    };
    auto sstore = [&](int buf) {
        #pragma unroll
        for (int i = 0; i < 4; ++i) {
            int idx = tid + (i << 8);
            int row = idx >> 3, cc = (idx & 7) << 2;
            if (A_BF16) {
                *(v2u*)&As[buf][row][cc] = aH[i];
            } else {
                v2u pk;
                pk[0] = (unsigned)f32_to_bf16(aF[i][0]) | ((unsigned)f32_to_bf16(aF[i][1]) << 16);
                pk[1] = (unsigned)f32_to_bf16(aF[i][2]) | ((unsigned)f32_to_bf16(aF[i][3]) << 16);
                *(v2u*)&As[buf][row][cc] = pk;
            }
            int krow = idx >> 5, cc2 = (idx & 31) << 2;
            Bts[buf][cc2 + 0][krow] = f32_to_bf16(wR[i][0]);
            Bts[buf][cc2 + 1][krow] = f32_to_bf16(wR[i][1]);
            Bts[buf][cc2 + 2][krow] = f32_to_bf16(wR[i][2]);
            Bts[buf][cc2 + 3][krow] = f32_to_bf16(wR[i][3]);
        }
    };

    gload(0);
    sstore(0);
    __syncthreads();

    const int nkt = K / TILE_K;
    for (int kt = 0; kt < nkt; ++kt) {
        const int cur = kt & 1;
        // issue next tile's global loads first (latency hidden by WMMAs below)
        if (kt + 1 < nkt) gload((kt + 1) * TILE_K);
        if (kt + 2 < nkt)                            // -> global_prefetch_b8
            __builtin_prefetch(&W[(size_t)(kt + 2) * TILE_K * N + n0 + (tid & 31) * 4], 0, 1);

        // ---- batched fragment loads, then 8 WMMAs ----
        v16bf a[2], bfr[4];
        #pragma unroll
        for (int mt = 0; mt < 2; ++mt) {
            int r = wm * 32 + mt * 16 + lr;
            a[mt] = load_frag(&As[cur][r][hk8], &As[cur][r][16 + hk8]);
        }
        #pragma unroll
        for (int nt = 0; nt < 4; ++nt) {
            int ccol = wn * 64 + nt * 16 + lr;
            bfr[nt] = load_frag(&Bts[cur][ccol][hk8], &Bts[cur][ccol][16 + hk8]);
        }
        #pragma unroll
        for (int nt = 0; nt < 4; ++nt)
            #pragma unroll
            for (int mt = 0; mt < 2; ++mt)
                acc[mt][nt] = wmma_bf16(a[mt], bfr[nt], acc[mt][nt]);

        // stage next tile into the other buffer, single barrier per K-step
        if (kt + 1 < nkt) sstore(cur ^ 1);
        __syncthreads();
    }

    // ---- epilogue: bias + store f32 ----
    #pragma unroll
    for (int mt = 0; mt < 2; ++mt) {
        int rbase = m0 + wm * 32 + mt * 16 + hk8;
        #pragma unroll
        for (int nt = 0; nt < 4; ++nt) {
            int col  = n0 + wn * 64 + nt * 16 + lr;
            float bv = bias ? bias[col] : 0.0f;
            #pragma unroll
            for (int j = 0; j < 8; ++j) {
                int r = rbase + j;
                if (r < M) Yb[(size_t)r * N + col] = acc[mt][nt][j] + bv;
            }
        }
    }
}

// ---------------------------------------------------------------------------
// Row finalize: optional RMS-norm (over 2048) * gain, f32 -> bf16
// ---------------------------------------------------------------------------
template <int NORM>
__global__ __launch_bounds__(256)
void finalize_rows(const float* __restrict__ Yin, const float* __restrict__ g,
                   unsigned short* __restrict__ Out)
{
    const int row = blockIdx.x;
    const int tid = threadIdx.x;
    const float* y      = Yin + (size_t)row * 2048;
    unsigned short* out = Out + (size_t)row * 2048;

    float vals[8];
    float ss = 0.0f;
    #pragma unroll
    for (int i = 0; i < 8; ++i) {
        vals[i] = y[tid + (i << 8)];
        ss += vals[i] * vals[i];
    }
    #pragma unroll
    for (int off = 16; off; off >>= 1) ss += __shfl_xor(ss, off, 32);
    __shared__ float red[8];
    if ((tid & 31) == 0) red[tid >> 5] = ss;
    __syncthreads();
    float tot = 0.0f;
    #pragma unroll
    for (int i = 0; i < 8; ++i) tot += red[i];
    const float scale = NORM ? rsqrtf(tot * (1.0f / 2048.0f) + 1e-6f) : 1.0f;
    #pragma unroll
    for (int i = 0; i < 8; ++i) {
        int c = tid + (i << 8);
        float v = NORM ? vals[i] * scale * g[c] : vals[i];
        out[c] = f32_to_bf16(v);
    }
}

// ---------------------------------------------------------------------------
// Fused dual cross-attention (img + masked txt), flash style, bf16 WMMA.
// Grid: (L1/128, NH, B). 8 waves x 16 queries each.
// K chunks: async global->LDS copy, double-buffered, one chunk in flight.
// V chunks: register staging (needs transpose + zero-fill).
// ---------------------------------------------------------------------------
__global__ __launch_bounds__(256)
void cross_attn(const unsigned short* __restrict__ Q,
                const unsigned short* __restrict__ Ktxt,
                const unsigned short* __restrict__ Vtxt,
                const unsigned short* __restrict__ Kimg,
                const unsigned short* __restrict__ Vimg,
                const int* __restrict__ ctx_lens,
                unsigned short* __restrict__ O)
{
    constexpr int L1 = 4096, C = 2048, HD = 128;
    const int tid  = threadIdx.x;
    const int lane = tid & 31;
    const int wave = tid >> 5;
    const int lr   = lane & 15;
    const int half = lane >> 4;
    const int hk8  = half << 3;

    const int h = blockIdx.y;
    const int b = blockIdx.z;
    const int qrow0 = blockIdx.x * 128 + wave * 16;

    __shared__ unsigned short Ks [2][64][HD + 8];    // double-buffered K chunk
    __shared__ unsigned short Vts[HD][64 + 8];       // transposed V chunk
    __shared__ unsigned short Ps [8][16][64 + 8];    // per-wave probs (A-layout)

    // ---- load Q fragments once (16 rows x 128 hd -> 4 k-steps of 32) ----
    v16bf qf[4];
    {
        const unsigned short* qrow = Q + ((size_t)(b * L1 + qrow0 + lr)) * C + h * HD;
        #pragma unroll
        for (int kk = 0; kk < 4; ++kk)
            qf[kk] = load_frag(&qrow[kk * 32 + hk8], &qrow[kk * 32 + 16 + hk8]);
    }

    v8f Ofin[8];
    #pragma unroll
    for (int nt = 0; nt < 8; ++nt) Ofin[nt] = (v8f)0.0f;

    int lt = ctx_lens[b];
    if (lt < 1) lt = 1;
    if (lt > 512) lt = 512;

    const float sc = 0.08838834764831845f * 1.4426950408889634f; // 1/sqrt(128)*log2(e)

    v4u vReg[4];   // V chunk staging registers (pipelined)

    #pragma unroll 1
    for (int pass = 0; pass < 2; ++pass) {
        const unsigned short* Kg = pass ? (Ktxt + (size_t)b * 512 * C)
                                        : (Kimg + (size_t)b * 257 * C);
        const unsigned short* Vg = pass ? (Vtxt + (size_t)b * 512 * C)
                                        : (Vimg + (size_t)b * 257 * C);
        const int nk      = pass ? 512 : 257;
        const int kvlen   = pass ? lt  : 257;
        const int nchunks = (nk + 63) >> 6;

        // async K chunk -> LDS (clamped row for tail; masked in softmax anyway)
        auto async_k = [&](int c, int buf) {
            #pragma unroll
            for (int i = 0; i < 4; ++i) {
                int idx = tid + (i << 8);            // 64 rows x 16 chunks of 8
                int kr = idx >> 4, cc = (idx & 15) << 3;
                int key = c * 64 + kr;
                if (key >= nk) key = nk - 1;
                async_copy_b128(&Kg[(size_t)key * C + h * HD + cc],
                                &Ks[buf][kr][cc]);
            }
        };
        auto load_v = [&](int c) {
            #pragma unroll
            for (int i = 0; i < 4; ++i) {
                int idx = tid + (i << 8);
                int kr = idx >> 4, cc = (idx & 15) << 3;
                int key = c * 64 + kr;
                vReg[i] = (v4u)0u;                   // zero-fill: no NaN into P@V
                if (key < nk)
                    vReg[i] = *(const v4u*)&Vg[(size_t)key * C + h * HD + cc];
            }
        };
        auto store_v = [&]() {
            #pragma unroll
            for (int i = 0; i < 4; ++i) {
                int idx = tid + (i << 8);
                int kr = idx >> 4, cc = (idx & 15) << 3;
                const unsigned short* e = (const unsigned short*)&vReg[i];
                #pragma unroll
                for (int q = 0; q < 8; ++q) Vts[cc + q][kr] = e[q];
            }
        };

        float rmax[8], rsum[8];
        v8f   Oacc[8];
        #pragma unroll
        for (int j = 0; j < 8; ++j) { rmax[j] = -3.0e38f; rsum[j] = 0.0f; }
        #pragma unroll
        for (int nt = 0; nt < 8; ++nt) Oacc[nt] = (v8f)0.0f;

        __syncthreads();          // previous pass's LDS reads fully retired
        async_k(0, 0);
        load_v(0);

        for (int c = 0; c < nchunks; ++c) {
            const int cur = c & 1;
            __syncthreads();                 // (1) prior chunk's LDS reads done
            store_v();                       // Vts <- regs (chunk c)
            if (c + 1 < nchunks) {
                async_k(c + 1, cur ^ 1);     // next K chunk in flight
                load_v(c + 1);               // next V chunk -> regs
                wait_async_le4();            // oldest 4 (chunk c) complete
            } else {
                wait_async_0();
            }
            __syncthreads();                 // (2) Ks[cur] + Vts visible to all

            // ---- scores: S(16x64) = Q(16x128) @ K^T, batched frag loads ----
            v8f S[4];
            #pragma unroll
            for (int nt = 0; nt < 4; ++nt) S[nt] = (v8f)0.0f;
            #pragma unroll
            for (int kk = 0; kk < 4; ++kk) {
                const int co = kk * 32 + hk8;
                v16bf kf[4];
                #pragma unroll
                for (int nt = 0; nt < 4; ++nt)
                    kf[nt] = load_frag(&Ks[cur][nt * 16 + lr][co],
                                       &Ks[cur][nt * 16 + lr][co + 16]);
                #pragma unroll
                for (int nt = 0; nt < 4; ++nt)
                    S[nt] = wmma_bf16(qf[kk], kf[nt], S[nt]);
            }

            // ---- online softmax (rows live on 16-lane halves) ----
            #pragma unroll
            for (int nt = 0; nt < 4; ++nt) {
                int key = c * 64 + nt * 16 + lr;
                bool ok = key < kvlen;
                #pragma unroll
                for (int j = 0; j < 8; ++j)
                    S[nt][j] = ok ? S[nt][j] * sc : -3.0e38f;
            }
            #pragma unroll
            for (int j = 0; j < 8; ++j) {
                float m = fmaxf(fmaxf(S[0][j], S[1][j]), fmaxf(S[2][j], S[3][j]));
                m = fmaxf(m, __shfl_xor(m, 1, 32));
                m = fmaxf(m, __shfl_xor(m, 2, 32));
                m = fmaxf(m, __shfl_xor(m, 4, 32));
                m = fmaxf(m, __shfl_xor(m, 8, 32));
                float mnew = fmaxf(rmax[j], m);
                float corr = exp2f(rmax[j] - mnew);
                rmax[j] = mnew;
                float s = 0.0f;
                #pragma unroll
                for (int nt = 0; nt < 4; ++nt) {
                    float p = exp2f(S[nt][j] - mnew);
                    S[nt][j] = p;
                    s += p;
                }
                s += __shfl_xor(s, 1, 32);
                s += __shfl_xor(s, 2, 32);
                s += __shfl_xor(s, 4, 32);
                s += __shfl_xor(s, 8, 32);
                rsum[j] = rsum[j] * corr + s;
                #pragma unroll
                for (int d = 0; d < 8; ++d) Oacc[d][j] *= corr;
            }

            // ---- re-layout P (C-layout f32 -> A-layout bf16) via LDS ----
            #pragma unroll
            for (int nt = 0; nt < 4; ++nt)
                #pragma unroll
                for (int j = 0; j < 8; ++j)
                    Ps[wave][hk8 + j][nt * 16 + lr] = f32_to_bf16(S[nt][j]);
            __syncthreads();

            // ---- O += P(16x64) @ V(64x128), V-frags in groups of 4 ----
            #pragma unroll
            for (int kk2 = 0; kk2 < 2; ++kk2) {
                const int po = kk2 * 32 + hk8;
                v16bf pf = load_frag(&Ps[wave][lr][po], &Ps[wave][lr][po + 16]);
                const int vo = kk2 * 32 + hk8;
                #pragma unroll
                for (int g4 = 0; g4 < 2; ++g4) {
                    v16bf vf[4];
                    #pragma unroll
                    for (int t = 0; t < 4; ++t) {
                        int nt = g4 * 4 + t;
                        vf[t] = load_frag(&Vts[nt * 16 + lr][vo],
                                          &Vts[nt * 16 + lr][vo + 16]);
                    }
                    #pragma unroll
                    for (int t = 0; t < 4; ++t)
                        Oacc[g4 * 4 + t] = wmma_bf16(pf, vf[t], Oacc[g4 * 4 + t]);
                }
            }
        } // chunks

        #pragma unroll
        for (int j = 0; j < 8; ++j) {
            float inv = 1.0f / rsum[j];
            #pragma unroll
            for (int nt = 0; nt < 8; ++nt)
                Ofin[nt][j] += Oacc[nt][j] * inv;
        }
    } // pass

    // ---- store bf16 attention output (b, l, h*128 + d) ----
    unsigned short* orow = O + ((size_t)(b * L1 + qrow0)) * C + h * HD;
    #pragma unroll
    for (int nt = 0; nt < 8; ++nt)
        #pragma unroll
        for (int j = 0; j < 8; ++j)
            orow[(size_t)(hk8 + j) * C + nt * 16 + lr] = f32_to_bf16(Ofin[nt][j]);
}

// ---------------------------------------------------------------------------
// Host launcher
// ---------------------------------------------------------------------------
extern "C" void kernel_launch(void* const* d_in, const int* in_sizes, int n_in,
                              void* d_out, int out_size, void* d_ws, size_t ws_size,
                              hipStream_t stream)
{
    (void)in_sizes; (void)n_in; (void)out_size; (void)ws_size;

    const float* x        = (const float*)d_in[0];
    const float* context  = (const float*)d_in[1];
    const int*   ctx_lens = (const int*)  d_in[2];
    const float* Wq  = (const float*)d_in[3];
    const float* bq  = (const float*)d_in[4];
    const float* gq  = (const float*)d_in[5];
    const float* Wk  = (const float*)d_in[6];
    const float* bk  = (const float*)d_in[7];
    const float* gk  = (const float*)d_in[8];
    const float* Wv  = (const float*)d_in[9];
    const float* bv  = (const float*)d_in[10];
    const float* Wki = (const float*)d_in[11];
    const float* bki = (const float*)d_in[12];
    const float* gki = (const float*)d_in[13];
    const float* Wvi = (const float*)d_in[14];
    const float* bvi = (const float*)d_in[15];
    const float* Wo  = (const float*)d_in[16];
    const float* bo  = (const float*)d_in[17];

    const int B = 2, L1 = 4096, C = 2048, L2 = 769, IMG = 257, TXT = 512;

    auto alignup = [](size_t v) { return (v + 255) & ~(size_t)255; };
    char* ws = (char*)d_ws;
    size_t off = 0;
    float*          Sbuf = (float*)(ws + off);          off += alignup((size_t)B * L1 * C * 4);
    unsigned short* qbf  = (unsigned short*)(ws + off); off += alignup((size_t)B * L1 * C * 2);
    unsigned short* kbf  = (unsigned short*)(ws + off); off += alignup((size_t)B * TXT * C * 2);
    unsigned short* vbf  = (unsigned short*)(ws + off); off += alignup((size_t)B * TXT * C * 2);
    unsigned short* kibf = (unsigned short*)(ws + off); off += alignup((size_t)B * IMG * C * 2);
    unsigned short* vibf = (unsigned short*)(ws + off); off += alignup((size_t)B * IMG * C * 2);
    unsigned short* abf  = (unsigned short*)(ws + off); off += alignup((size_t)B * L1 * C * 2);

    const dim3 blk(256);
    const float* ctx_txt = context + (size_t)IMG * C;

    // Q = rms(x @ Wq + bq) * gq
    gemm_bias<0><<<dim3(16, 32, 2), blk, 0, stream>>>(x, Wq, bq, Sbuf,
        L1, C, C, (long long)L1 * C, (long long)L1 * C);
    finalize_rows<1><<<B * L1, blk, 0, stream>>>(Sbuf, gq, qbf);

    // K_txt = rms(ctx_txt @ Wk + bk) * gk
    gemm_bias<0><<<dim3(16, 4, 2), blk, 0, stream>>>(ctx_txt, Wk, bk, Sbuf,
        TXT, C, C, (long long)L2 * C, (long long)TXT * C);
    finalize_rows<1><<<B * TXT, blk, 0, stream>>>(Sbuf, gk, kbf);

    // V_txt = ctx_txt @ Wv + bv
    gemm_bias<0><<<dim3(16, 4, 2), blk, 0, stream>>>(ctx_txt, Wv, bv, Sbuf,
        TXT, C, C, (long long)L2 * C, (long long)TXT * C);
    finalize_rows<0><<<B * TXT, blk, 0, stream>>>(Sbuf, nullptr, vbf);

    // K_img = rms(ctx_img @ Wki + bki) * gki
    gemm_bias<0><<<dim3(16, 3, 2), blk, 0, stream>>>(context, Wki, bki, Sbuf,
        IMG, C, C, (long long)L2 * C, (long long)IMG * C);
    finalize_rows<1><<<B * IMG, blk, 0, stream>>>(Sbuf, gki, kibf);

    // V_img = ctx_img @ Wvi + bvi
    gemm_bias<0><<<dim3(16, 3, 2), blk, 0, stream>>>(context, Wvi, bvi, Sbuf,
        IMG, C, C, (long long)L2 * C, (long long)IMG * C);
    finalize_rows<0><<<B * IMG, blk, 0, stream>>>(Sbuf, nullptr, vibf);

    // fused img + masked-txt attention
    cross_attn<<<dim3(32, 16, 2), blk, 0, stream>>>(qbf, kbf, vbf, kibf, vibf,
                                                    ctx_lens, abf);

    // out = attn @ Wo + bo   (f32 output)
    gemm_bias<1><<<dim3(16, 32, 2), blk, 0, stream>>>(abf, Wo, bo, (float*)d_out,
        L1, C, C, (long long)L1 * C, (long long)L1 * C);
}
